// CARAFEUpsample_62843961475319
// MI455X (gfx1250) — matrix operand order repaired and checked by reading, not compiled, not run on hardware
//
#include <hip/hip_runtime.h>
#include <hip/hip_bf16.h>

// ---------------------------------------------------------------------------
// CARAFE upsample on gfx1250 (MI455X): bf16 WMMA GEMMs + VALU reassembly.
// Shapes: B=2, Cin=512, Cout=256, Ccomp=64, Cmask=100 (k=5,s=2,g=1), H=W=64.
//
// All three conv stages run as barrier-free WMMA GEMMs with activations
// pre-packed into per-lane WMMA B-fragment order:
//   packed[...][kstep][lane][t],  lane = (kin>=16)*16 + n%16, t = kin%16
// Inner loops are pure {global_load_b128, v_wmma_f32_16x16x32_bf16}.
// ---------------------------------------------------------------------------

typedef __attribute__((ext_vector_type(16))) __bf16 v16bf;
typedef __attribute__((ext_vector_type(8)))  __bf16 v8bf;
typedef __attribute__((ext_vector_type(8)))  float  v8f;

#define HW        4096   // 64*64
#define NTILES_PB 256    // HW/16

// ---------------- conversion / packing kernels ------------------------------
__global__ void cvt_f32_to_bf16(const float* __restrict__ src,
                                __bf16* __restrict__ dst, int n) {
  int i = blockIdx.x * 256 + threadIdx.x;
  if (i < n) dst[i] = (__bf16)src[i];
}

// ce_w [100][64][3][3] f32 -> [9][112][64] bf16 (rows 100..111 zero)
__global__ void pack_cew(const float* __restrict__ cew, __bf16* __restrict__ dst) {
  int i = blockIdx.x * 256 + threadIdx.x;
  if (i >= 9 * 112 * 64) return;
  int cin = i & 63;
  int t   = i >> 6;
  int m   = t % 112;
  int pos = t / 112;
  __bf16 v = (__bf16)0.f;
  if (m < 100) v = (__bf16)cew[(m * 64 + cin) * 9 + pos];
  dst[i] = v;
}

// x [B][K][4096] f32 -> packed bf16 [B*256 ntile][K/32][32 lane][16 t]
__global__ void pack_x_bf16(const float* __restrict__ x,
                            __bf16* __restrict__ xp, int Kdim) {
  int i = blockIdx.x * 256 + threadIdx.x;
  int total = 2 * Kdim * 256;
  if (i >= total) return;
  int ntile = i & 255;
  int k     = (i >> 8) % Kdim;
  int batch = i / (Kdim * 256);
  const float* src = x + ((size_t)batch * Kdim + k) * HW + ntile * 16;
  int kstep = k >> 5, kin = k & 31, half = kin >> 4, t = kin & 15;
  __bf16* dst = xp + ((((size_t)(batch * 256 + ntile) * (Kdim / 32) + kstep) * 32
                       + half * 16) * 16) + t;
#pragma unroll
  for (int nl = 0; nl < 16; ++nl) dst[nl * 16] = (__bf16)src[nl];
}

// comp_bf [B][64][4096] -> 9 shifted zero-padded packed operands
// comp9[pos][batch*256+ntile][kstep(2)][32 lane][16 t]
__global__ void pack_comp9(const __bf16* __restrict__ comp,
                           __bf16* __restrict__ pk9) {
  int i = blockIdx.x * 256 + threadIdx.x;        // 9*2*64*256
  if (i >= 9 * 2 * 64 * 256) return;
  int ntile = i & 255;
  int k     = (i >> 8) & 63;
  int batch = (i >> 14) & 1;
  int pos   = i >> 15;                            // 0..8
  int dy = pos / 3 - 1, dx = pos % 3 - 1;
  int h0 = ntile >> 2;                            // n0/64
  int w0 = (ntile & 3) << 4;
  int hh = h0 + dy;
  const __bf16* src = comp + ((size_t)batch * 64 + k) * HW + hh * 64;
  int kstep = k >> 5, kin = k & 31, half = kin >> 4, t = kin & 15;
  __bf16* dst = pk9 + ((((size_t)pos * 512 + batch * 256 + ntile) * 2 + kstep) * 32
                       + half * 16) * 16 + t;
  bool hok = (unsigned)hh < 64u;
#pragma unroll
  for (int nl = 0; nl < 16; ++nl) {
    int ww = w0 + nl + dx;
    __bf16 v = (__bf16)0.f;
    if (hok && (unsigned)ww < 64u) v = src[ww];
    dst[nl * 16] = v;
  }
}

// ---------------- barrier-free packed WMMA GEMM (1x1 convs) -----------------
// Out[b][M][4096] = A[M,K] @ In + bias ; wave computes 64M x 16N strip.
template <int K>
__global__ void wmma_gemm_packed(const __bf16* __restrict__ A,
                                 const __bf16* __restrict__ Bp,
                                 const float* __restrict__ bias,
                                 float* __restrict__ outF,
                                 __bf16* __restrict__ outBFstd,
                                 __bf16* __restrict__ outBFpk,
                                 int M, int mgroups) {
  constexpr int KSTEPS = K / 32;
  const int tid  = threadIdx.x;
  const int wave = tid >> 5;
  const int lane = tid & 31;
  const int nl   = lane & 15;
  const int half = lane >> 4;

  const int wg      = blockIdx.x * 8 + wave;
  const int ntile_g = wg / mgroups;          // 0 .. batch*256-1
  const int mg      = wg % mgroups;
  const int batch   = ntile_g >> 8;
  const int ntile   = ntile_g & 255;
  const int n0      = ntile << 4;

  const __bf16* Bt = Bp + (size_t)(batch * 256 + ntile) * KSTEPS * 512 + lane * 16;
  const __bf16* Ar = A + (size_t)(mg * 64 + nl) * K;

  v8f acc[4] = {};
#pragma unroll
  for (int ks = 0; ks < KSTEPS; ++ks) {
    union { v16bf v; v8bf h[2]; } b;
    b.v = *(const v16bf*)(Bt + (size_t)ks * 512);
#pragma unroll
    for (int mt = 0; mt < 4; ++mt) {
      union { v16bf v; v8bf h[2]; } a;
      const __bf16* ar = Ar + (size_t)mt * 16 * K + ks * 32;
      a.h[0] = *(const v8bf*)(ar + half * 16);
      a.h[1] = *(const v8bf*)(ar + half * 16 + 8);
      acc[mt] = __builtin_amdgcn_wmma_f32_16x16x32_bf16(
          false, a.v, false, b.v, (short)0, acc[mt], false, false);
    }
  }

#pragma unroll
  for (int mt = 0; mt < 4; ++mt) {
#pragma unroll
    for (int v = 0; v < 8; ++v) {
      int m = mg * 64 + mt * 16 + v + half * 8;
      float r = acc[mt][v] + bias[m];
      size_t soff = (size_t)batch * M * HW + (size_t)m * HW + n0 + nl;
      if (outF)     outF[soff]     = r;
      if (outBFstd) outBFstd[soff] = (__bf16)r;
      if (outBFpk) {                       // packed operand for the next GEMM (K2=M)
        int kstep2 = m >> 5, kin = m & 31, h2 = kin >> 4, t2 = kin & 15;
        size_t idx = (((size_t)(batch * 256 + ntile) * (M / 32) + kstep2) * 32
                      + h2 * 16 + nl) * 16 + t2;
        outBFpk[idx] = (__bf16)r;
      }
    }
  }
}

// ---------------- 3x3 conv as barrier-free packed WMMA GEMM -----------------
// mask[b][100][4096]; A = W9[9][112][64]; B = comp9 (9 shifted packed ops).
// Wave owns one 16-col tile and all 7 M-tiles. 126 WMMAs/wave, no LDS.
__global__ void conv3x3_wmma_packed(const __bf16* __restrict__ W9,
                                    const __bf16* __restrict__ pk9,
                                    const float* __restrict__ bias,
                                    float* __restrict__ mask) {
  const int tid  = threadIdx.x;
  const int wave = tid >> 5;
  const int lane = tid & 31;
  const int nl   = lane & 15;
  const int half = lane >> 4;

  const int wg    = blockIdx.x * 8 + wave;   // 0..511
  const int batch = wg >> 8;
  const int ntile = wg & 255;
  const int n0    = ntile << 4;

  v8f acc[7] = {};
  for (int pos = 0; pos < 9; ++pos) {
    const __bf16* Bt = pk9 + (((size_t)pos * 512 + batch * 256 + ntile) * 2) * 512
                       + lane * 16;
    const __bf16* Abase = W9 + (size_t)pos * 112 * 64;
#pragma unroll
    for (int ks = 0; ks < 2; ++ks) {
      union { v16bf v; v8bf h[2]; } b;
      b.v = *(const v16bf*)(Bt + (size_t)ks * 512);
#pragma unroll
      for (int mt = 0; mt < 7; ++mt) {
        union { v16bf v; v8bf h[2]; } a;
        const __bf16* ar = Abase + (size_t)(mt * 16 + nl) * 64 + ks * 32;
        a.h[0] = *(const v8bf*)(ar + half * 16);
        a.h[1] = *(const v8bf*)(ar + half * 16 + 8);
        acc[mt] = __builtin_amdgcn_wmma_f32_16x16x32_bf16(
            false, a.v, false, b.v, (short)0, acc[mt], false, false);
      }
    }
  }

  float* mb = mask + (size_t)batch * 100 * HW;
#pragma unroll
  for (int mt = 0; mt < 7; ++mt) {
#pragma unroll
    for (int v = 0; v < 8; ++v) {
      int m = mt * 16 + v + half * 8;
      if (m < 100) mb[(size_t)m * HW + n0 + nl] = acc[mt][v] + bias[m];
    }
  }
}

// ---------------- pixel-shuffle + softmax(25) + 25-tap reassembly -----------
__global__ void carafe_reassembly(const float* __restrict__ feat,
                                  const float* __restrict__ mask,
                                  float* __restrict__ out) {
  __shared__ float lg[16][4][25];
  __shared__ float wt[16][4][25];
  const int tid   = threadIdx.x;
  const int tile  = blockIdx.x;
  const int batch = tile >> 8;
  const int hw0   = (tile & 255) << 4;
  const int h     = hw0 >> 6;
  const int wb    = hw0 & 63;

  const float* mb = mask + (size_t)batch * 100 * HW;
  for (int idx = tid; idx < 16 * 100; idx += 256) {
    int pix = idx / 100, r = idx % 100;
    int sp = r / 25, kk = r % 25;
    lg[pix][sp][kk] = mb[(size_t)(kk * 4 + sp) * HW + hw0 + pix];
  }
  __syncthreads();

  if (tid < 64) {
    int pix = tid >> 2, sp = tid & 3;
    float mx = -1e30f;
    for (int kk = 0; kk < 25; ++kk) mx = fmaxf(mx, lg[pix][sp][kk]);
    float s = 0.f, e[25];
    for (int kk = 0; kk < 25; ++kk) { e[kk] = __expf(lg[pix][sp][kk] - mx); s += e[kk]; }
    float inv = 1.f / s;
    for (int kk = 0; kk < 25; ++kk) wt[pix][sp][kk] = e[kk] * inv;
  }
  __syncthreads();

  const int pix   = tid & 15;        // w offset within tile (coalesced)
  const int cslot = tid >> 4;        // channel slot
  const int w     = wb + pix;
  const float* fb = feat + (size_t)batch * 256 * HW;
  float*       ob = out  + (size_t)batch * 256 * 16384;

  for (int ci = 0; ci < 16; ++ci) {
    int c = cslot + ci * 16;
    const float* fc = fb + (size_t)c * HW;
    float a0 = 0.f, a1 = 0.f, a2 = 0.f, a3 = 0.f;
#pragma unroll
    for (int kk = 0; kk < 25; ++kk) {
      int dy = kk / 5 - 2, dx = kk % 5 - 2;
      int hh = h + dy, ww = w + dx;
      float f = 0.f;
      if ((unsigned)hh < 64u && (unsigned)ww < 64u) f = fc[hh * 64 + ww];
      a0 = fmaf(wt[pix][0][kk], f, a0);
      a1 = fmaf(wt[pix][1][kk], f, a1);
      a2 = fmaf(wt[pix][2][kk], f, a2);
      a3 = fmaf(wt[pix][3][kk], f, a3);
    }
    float* oc = ob + (size_t)c * 16384 + (2 * h) * 128 + 2 * w;
    oc[0] = a0; oc[1] = a1; oc[128] = a2; oc[129] = a3;
  }
}

// ---------------------------------------------------------------------------
extern "C" void kernel_launch(void* const* d_in, const int* in_sizes, int n_in,
                              void* d_out, int out_size, void* d_ws, size_t ws_size,
                              hipStream_t stream) {
  const float* x      = (const float*)d_in[0];   // [2,512,64,64]
  const float* enc_w  = (const float*)d_in[1];   // [256,512,1,1]
  const float* enc_b  = (const float*)d_in[2];   // [256]
  const float* comp_w = (const float*)d_in[3];   // [64,256,1,1]
  const float* comp_b = (const float*)d_in[4];   // [64]
  const float* ce_w   = (const float*)d_in[5];   // [100,64,3,3]
  const float* ce_b   = (const float*)d_in[6];   // [100]
  float* out = (float*)d_out;                    // [2,256,128,128]

  char* ws = (char*)d_ws;
  __bf16* xp      = (__bf16*)(ws);                    // packed x: 8 MB
  __bf16* ewb     = (__bf16*)(ws + 8388608);          // 256*512
  __bf16* cwb     = (__bf16*)(ws + 8650752);          // 64*256
  __bf16* cewb    = (__bf16*)(ws + 8683520);          // 9*112*64
  float*  feat    = (float*) (ws + 8812544);          // 2*256*4096 f32 = 8 MB
  __bf16* feat_pk = (__bf16*)(ws + 17201152);         // packed feat: 4 MB
  __bf16* comp_bf = (__bf16*)(ws + 21395456);         // 2*64*4096 bf16
  float*  maskbuf = (float*) (ws + 22444032);         // 2*100*4096 f32
  __bf16* comp9   = (__bf16*)(ws + 25720832);         // 9 packed shifted: 9.4 MB

  // 1) precision conversion / packing
  pack_x_bf16<<<(2 * 512 * 256 + 255) / 256, 256, 0, stream>>>(x, xp, 512);
  cvt_f32_to_bf16<<<(131072 + 255) / 256, 256, 0, stream>>>(enc_w, ewb, 131072);
  cvt_f32_to_bf16<<<(16384 + 255) / 256, 256, 0, stream>>>(comp_w, cwb, 16384);
  pack_cew<<<(9 * 112 * 64 + 255) / 256, 256, 0, stream>>>(ce_w, cewb);

  // 2) encoder 1x1 conv: feat (f32) + packed bf16 operand for GEMM3
  wmma_gemm_packed<512><<<256, 256, 0, stream>>>(
      ewb, xp, enc_b, feat, (__bf16*)nullptr, feat_pk, 256, 4);

  // 3) channel compressor 1x1 conv: comp bf16 (std layout)
  wmma_gemm_packed<256><<<64, 256, 0, stream>>>(
      cwb, feat_pk, comp_b, (float*)nullptr, comp_bf, (__bf16*)nullptr, 64, 1);

  // 4) pack comp into 9 shifted zero-padded WMMA operands
  pack_comp9<<<(9 * 2 * 64 * 256 + 255) / 256, 256, 0, stream>>>(comp_bf, comp9);

  // 5) content encoder 3x3 conv -> mask logits [2,100,4096] (pure WMMA)
  conv3x3_wmma_packed<<<64, 256, 0, stream>>>(cewb, comp9, ce_b, maskbuf);

  // 6) shuffle + softmax + reassembly -> out [2,256,128,128]
  carafe_reassembly<<<2 * NTILES_PB, 256, 0, stream>>>(feat, maskbuf, out);
}
